// RelationalSAGE_16423954940680
// MI455X (gfx1250) — compile-verified, matrix operand order checked by reading, not run on hardware
//
#include <hip/hip_runtime.h>

// ---------------------------------------------------------------------------
// RelationalSAGE on MI455X (gfx1250).
// - All GEMMs run on v_wmma_f32_16x16x32_bf16 (wave32, 16x128 stripe/wave).
// - Packed weights staged in LDS once per workgroup (32KB of the WGP's 320KB).
// - B fragments double-buffered: ds_load_b128 pair for tile nt+1 issued
//   before the WMMA of tile nt -> LDS latency hidden behind matrix ops.
// - Activations used as GEMM-A keep a bf16 row-major shadow, written once
//   when finalized -> K-loop does two b128 loads per A fragment, no cvt VALU.
// - proj / T exist only in bf16 (halves scatter gather traffic).
// - Edge scatter: bf16 gather (8B/lane) + 4x f32 atomics.
// ---------------------------------------------------------------------------

typedef __attribute__((ext_vector_type(16))) __bf16 v16bf;
typedef __attribute__((ext_vector_type(8)))  float  v8f;

#define DD    128
#define NBV   200000
#define NB    80000
#define NLT   30000
#define NLEG  8000

__device__ __forceinline__ unsigned short f2bf(float x) {
  unsigned u = __float_as_uint(x);
  return (unsigned short)((u + 0x7FFFu + ((u >> 16) & 1u)) >> 16);
}
__device__ __forceinline__ float bf2f(unsigned short u) {
  return __uint_as_float(((unsigned)u) << 16);
}

// ---------------------------- utility kernels ------------------------------

__global__ void fill_zero_kernel(float* __restrict__ p, long n) {
  long i = (long)blockIdx.x * blockDim.x + threadIdx.x;
  if (i < n) p[i] = 0.0f;
}

// f32 -> bf16 shadow, 4 elements/thread
__global__ void cvt_bf16_kernel(const float* __restrict__ src,
                                unsigned short* __restrict__ dst, long n4) {
  long i = (long)blockIdx.x * blockDim.x + threadIdx.x;
  if (i >= n4) return;
  float4 x = ((const float4*)src)[i];
  ushort4 o;
  o.x = f2bf(x.x); o.y = f2bf(x.y); o.z = f2bf(x.z); o.w = f2bf(x.w);
  ((ushort4*)dst)[i] = o;
}

// Pack row-major f32 W[k*128+n] (K x 128) into per-lane WMMA B-fragment order
// (16-bit B layout, §7.12.2): packed[((kt*8+nt)*32+lane)*16+j]
__global__ void pack_w_kernel(const float* __restrict__ W,
                              const float* __restrict__ W2,
                              unsigned short* __restrict__ out, int K) {
  int idx = blockIdx.x * blockDim.x + threadIdx.x;
  if (idx >= K * 128) return;
  int j    = idx & 15;
  int lane = (idx >> 4) & 31;
  int nt   = (idx >> 9) & 7;
  int kt   = idx >> 12;
  int n  = nt * 16 + (lane & 15);
  int hi = lane >> 4;
  int k  = kt * 32 + ((j < 8) ? (hi * 8 + j) : (16 + hi * 8 + (j - 8)));
  float v = W[k * 128 + n];
  if (W2) v += W2[k * 128 + n];
  out[idx] = f2bf(v);
}

__global__ void deg_kernel(const int* __restrict__ dst, int E, float* __restrict__ deg) {
  int i = blockIdx.x * blockDim.x + threadIdx.x;
  if (i < E) atomicAdd(&deg[dst[i]], 1.0f);
}

// one thread = (edge, 4 dims): 8B bf16 gather + 4 f32 atomic adds
__global__ void scatter_kernel(const unsigned short* __restrict__ proj,
                               const int* __restrict__ src,
                               const int* __restrict__ dst,
                               long E, float* __restrict__ agg) {
  long idx = (long)blockIdx.x * blockDim.x + threadIdx.x;
  if (idx >= E * 32) return;
  long e  = idx >> 5;
  int  d4 = (int)(idx & 31) << 2;
  ushort4 p = *(const ushort4*)(proj + (long)src[e] * DD + d4);
  float* a = agg + (long)dst[e] * DD + d4;
  atomicAdd(a + 0, bf2f(p.x));
  atomicAdd(a + 1, bf2f(p.y));
  atomicAdd(a + 2, bf2f(p.z));
  atomicAdd(a + 3, bf2f(p.w));
}

// agg = agg/deg + b_src ; dual-write f32 + bf16 shadow
__global__ void normalize_kernel(float* __restrict__ agg,
                                 unsigned short* __restrict__ agg_bf,
                                 const float* __restrict__ deg,
                                 const float* __restrict__ bias, long M) {
  long idx = (long)blockIdx.x * blockDim.x + threadIdx.x;
  if (idx >= M * 32) return;
  long r  = idx >> 5;
  int  d4 = (int)(idx & 31) << 2;
  float dinv = 1.0f / fmaxf(deg[r], 1.0f);
  float4 a = ((const float4*)agg)[idx];
  float4 b = *(const float4*)(bias + d4);
  a.x = a.x * dinv + b.x; a.y = a.y * dinv + b.y;
  a.z = a.z * dinv + b.z; a.w = a.w * dinv + b.w;
  ((float4*)agg)[idx] = a;
  ushort4 o; o.x = f2bf(a.x); o.y = f2bf(a.y); o.z = f2bf(a.z); o.w = f2bf(a.w);
  ((ushort4*)agg_bf)[idx] = o;
}

// dst = src[rows[r]] ; dual-write f32 + bf16 shadow
__global__ void gather_rows_kernel(const float* __restrict__ src,
                                   const int* __restrict__ rows,
                                   float* __restrict__ dstf,
                                   unsigned short* __restrict__ dstb, long M) {
  long idx = (long)blockIdx.x * blockDim.x + threadIdx.x;
  if (idx >= M * 32) return;
  long r  = idx >> 5;
  int  d4 = (int)(idx & 31) << 2;
  float4 x = *(const float4*)(src + (long)rows[r] * DD + d4);
  ((float4*)dstf)[idx] = x;
  ushort4 o; o.x = f2bf(x.x); o.y = f2bf(x.y); o.z = f2bf(x.z); o.w = f2bf(x.w);
  ((ushort4*)dstb)[idx] = o;
}

// per-row: a = sigmoid(dot(Q,kk)/sqrt(128)); ht = 0.5*ht + 0.5*a*vv
// one wave/row; dual-writes bf16 shadow of the result
__global__ void attn_kernel(const float* __restrict__ Q,
                            const float* __restrict__ kk,
                            const float* __restrict__ vv,
                            float* __restrict__ ht,
                            unsigned short* __restrict__ ht_bf, int M) {
  int gid  = blockIdx.x * blockDim.x + threadIdx.x;
  int row  = gid >> 5;
  int lane = threadIdx.x & 31;
  if (row >= M) return;
  long base = (long)row * DD + lane * 4;
  float4 qv = *(const float4*)(Q + base);
  float4 kv = *(const float4*)(kk + base);
  float s = qv.x * kv.x + qv.y * kv.y + qv.z * kv.z + qv.w * kv.w;
  #pragma unroll
  for (int off = 16; off > 0; off >>= 1) s += __shfl_xor(s, off, 32);
  float a = 1.0f / (1.0f + __expf(-s / 11.3137084990f)); // sqrt(128)
  float4 vvv = *(const float4*)(vv + base);
  float4 hv  = *(const float4*)(ht + base);
  float4 o;
  o.x = 0.5f * hv.x + 0.5f * a * vvv.x;
  o.y = 0.5f * hv.y + 0.5f * a * vvv.y;
  o.z = 0.5f * hv.z + 0.5f * a * vvv.z;
  o.w = 0.5f * hv.w + 0.5f * a * vvv.w;
  *(float4*)(ht + base) = o;
  ushort4 ob; ob.x = f2bf(o.x); ob.y = f2bf(o.y); ob.z = f2bf(o.z); ob.w = f2bf(o.w);
  *(ushort4*)(ht_bf + base) = ob;
}

// g[row] = sigmoid(dot(H[row], w) + b2[0])   (one wave/row)
__global__ void rowdot_sig_kernel(const float* __restrict__ H,
                                  const float* __restrict__ w,
                                  const float* __restrict__ b2,
                                  float* __restrict__ g, int M) {
  int gid  = blockIdx.x * blockDim.x + threadIdx.x;
  int row  = gid >> 5;
  int lane = threadIdx.x & 31;
  if (row >= M) return;
  long base = (long)row * DD + lane * 4;
  float4 hv = *(const float4*)(H + base);
  float4 wv = *(const float4*)(w + lane * 4);
  float s = hv.x * wv.x + hv.y * wv.y + hv.z * wv.z + hv.w * wv.w;
  #pragma unroll
  for (int off = 16; off > 0; off >>= 1) s += __shfl_xor(s, off, 32);
  if (lane == 0) g[row] = 1.0f / (1.0f + __expf(-(s + b2[0])));
}

__global__ void gateout_kernel(const float* __restrict__ g,
                               const float* __restrict__ agg,
                               const float* __restrict__ bv,
                               float* __restrict__ out, long M) {
  long idx = (long)blockIdx.x * blockDim.x + threadIdx.x;
  if (idx >= M * 32) return;
  long r = idx >> 5;
  float gg = g[r];
  float4 a = ((const float4*)agg)[idx];
  float4 h = ((const float4*)bv)[idx];
  float4 o;
  o.x = gg * a.x + (1.0f - gg) * h.x;
  o.y = gg * a.y + (1.0f - gg) * h.y;
  o.z = gg * a.z + (1.0f - gg) * h.z;
  o.w = gg * a.w + (1.0f - gg) * h.w;
  ((float4*)out)[idx] = o;
}

// ------------------------------ bf16 WMMA GEMM -----------------------------
// C[M x 128] = epilogue( A_bf16[M x K] @ W[K x 128] )
// Block: 256 threads = 8 waves. Packed W staged cooperatively into LDS,
// then each wave computes a 16-row x 128-col stripe: 8 N-tiles x (K/32)
// WMMA steps. B fragments are double-buffered (loads for nt+1 issued before
// the WMMA of nt). Epilogue: +bias, +cadd, relu, *sigmoid(scale_ptr[idx]);
// writes f32 (=/+=) and/or a bf16 shadow.
template <int K>
__global__ __launch_bounds__(256) void gemm_bf16_kernel(
    const unsigned short* __restrict__ A, int lda, const int* __restrict__ a_rows,
    const unsigned short* __restrict__ Wp,
    const float* __restrict__ bias,
    const float* __restrict__ cadd,
    const float* __restrict__ scale_ptr, int scale_idx,
    int relu_flag, int accum_flag,
    float* __restrict__ Cf, unsigned short* __restrict__ Cbf, int M) {
  __shared__ unsigned short sW[K * 128];

  // cooperative stage of packed W into LDS (uint4 = 8 ushorts)
  {
    const uint4* gsrc = (const uint4*)Wp;
    uint4* ldst = (uint4*)sW;
    const int nvec = (K * 128) / 8;
    #pragma unroll
    for (int i = threadIdx.x; i < nvec; i += 256) ldst[i] = gsrc[i];
  }
  __syncthreads();

  const int lane = threadIdx.x & 31;
  const int wave = threadIdx.x >> 5;
  const int tileBase = (blockIdx.x * 8 + wave) * 16;
  if (tileBase >= M) return;             // after the barrier: tail waves idle
  const int hi = lane >> 4;
  const int ml = lane & 15;

  int arow = tileBase + ml;
  if (arow >= M) arow = M - 1;           // clamp; stores are guarded
  long asrc = a_rows ? (long)a_rows[arow] : (long)arow;
  const unsigned short* Ar = A + asrc * (long)lda;

  v8f acc[8];
  #pragma unroll
  for (int i = 0; i < 8; ++i)
    #pragma unroll
    for (int jj = 0; jj < 8; ++jj) acc[i][jj] = 0.0f;

  union BF { v16bf v; uint4 q[2]; };
  const int NKT = K / 32;

  // prime the B pipeline
  BF bcur;
  {
    const uint4* bp = (const uint4*)(sW + (long)lane * 16);
    bcur.q[0] = bp[0];
    bcur.q[1] = bp[1];
  }

  #pragma unroll
  for (int kt = 0; kt < NKT; ++kt) {
    // 16-bit A 16x32 layout (§7.12.2): lane holds A[m][kt*32+half*8 ..+7]
    // and A[m][kt*32+16+half*8 ..+7] -> two 16B global loads.
    union { v16bf v; uint4 q[2]; } af;
    const int k0 = kt * 32 + hi * 8;
    af.q[0] = *(const uint4*)(Ar + k0);
    af.q[1] = *(const uint4*)(Ar + k0 + 16);

    #pragma unroll
    for (int nt = 0; nt < 8; ++nt) {
      BF bnext = bcur;
      if (!(kt == NKT - 1 && nt == 7)) {
        const int fi = kt * 8 + nt + 1;  // next fragment (compile-time)
        const uint4* bp = (const uint4*)(sW + ((long)fi * 32 + lane) * 16);
        bnext.q[0] = bp[0];
        bnext.q[1] = bp[1];
      }
      acc[nt] = __builtin_amdgcn_wmma_f32_16x16x32_bf16(
          false, af.v, false, bcur.v, (short)0, acc[nt], false, false);
      bcur = bnext;
    }
  }

  float s = 1.0f;
  if (scale_ptr) s = 1.0f / (1.0f + __expf(-scale_ptr[scale_idx]));

  #pragma unroll
  for (int nt = 0; nt < 8; ++nt) {
    int col = nt * 16 + ml;
    float bia = bias ? bias[col] : 0.0f;
    #pragma unroll
    for (int v = 0; v < 8; ++v) {
      int row = tileBase + hi * 8 + v;   // C/D layout: lanes16-31 -> M+8
      if (row < M) {
        long o = (long)row * DD + col;
        float y = acc[nt][v] + bia;
        if (cadd) y += cadd[o];
        if (relu_flag) y = fmaxf(y, 0.0f);
        y *= s;
        if (Cf)  { if (accum_flag) Cf[o] += y; else Cf[o] = y; }
        if (Cbf) Cbf[o] = f2bf(y);
      }
    }
  }
}

// ------------------------------- host side ---------------------------------

static void launch_gemm(hipStream_t st, const unsigned short* A, int lda,
                        const int* arows, const unsigned short* Wp, int K,
                        const float* bias, const float* cadd,
                        const float* sp, int sidx, int relu, int accum,
                        float* Cf, unsigned short* Cbf, int M) {
  int blocks = (M + 127) / 128;
  if (K == 64)
    gemm_bf16_kernel<64><<<blocks, 256, 0, st>>>(A, lda, arows, Wp, bias, cadd,
                                                 sp, sidx, relu, accum, Cf, Cbf, M);
  else
    gemm_bf16_kernel<128><<<blocks, 256, 0, st>>>(A, lda, arows, Wp, bias, cadd,
                                                  sp, sidx, relu, accum, Cf, Cbf, M);
}

extern "C" void kernel_launch(void* const* d_in, const int* in_sizes, int n_in,
                              void* d_out, int out_size, void* d_ws, size_t ws_size,
                              hipStream_t stream) {
  (void)in_sizes; (void)n_in; (void)out_size; (void)ws_size;

  const float* in_h[4] = { (const float*)d_in[0], (const float*)d_in[1],
                           (const float*)d_in[2], (const float*)d_in[3] };
  const int* ev  = (const int*)d_in[4];
  const int* ei  = (const int*)d_in[5];
  const int* es  = (const int*)d_in[6];
  const int* esp = (const int*)d_in[7];
  const int* topic_ids = (const int*)d_in[8];
  const float* W_src = (const float*)d_in[9];
  const float* b_src = (const float*)d_in[10];
  const float* W_dst = (const float*)d_in[11];
  const float* b_dst = (const float*)d_in[12];
  const float* W_m   = (const float*)d_in[13];
  const float* b_m   = (const float*)d_in[14];
  const float* rel_attn = (const float*)d_in[15];
  const float* gW1 = (const float*)d_in[16];
  const float* gb1 = (const float*)d_in[17];
  const float* gW2 = (const float*)d_in[18];
  const float* gb2 = (const float*)d_in[19];
  const float* topic_table = (const float*)d_in[20];
  const float* W_tp = (const float*)d_in[21];
  const float* b_tp = (const float*)d_in[22];
  const float* W_q = (const float*)d_in[23];
  const float* b_q = (const float*)d_in[24];
  const float* W_k = (const float*)d_in[25];
  const float* b_k = (const float*)d_in[26];
  const float* W_v = (const float*)d_in[27];
  const float* b_v = (const float*)d_in[28];

  float* out     = (float*)d_out;
  float* out_bv  = out;
  float* out_b   = out_bv + (long)NBV * DD;
  float* out_lt  = out_b  + (long)NB  * DD;
  float* out_leg = out_lt + (long)NLT * DD;

  char* wsb = (char*)d_ws;
  size_t off = 0;
  auto alloc = [&](size_t bytes) -> void* {
    size_t a = (off + 255) & ~(size_t)255;
    off = a + bytes;
    return (void*)(wsb + a);
  };
  const long Nn[4] = { NBV, NB, NLT, NLEG };

  float* new0[4];
  for (int t = 0; t < 4; ++t) new0[t] = (float*)alloc((size_t)Nn[t] * DD * 4);
  float* new1_bv = (float*)alloc((size_t)NBV * DD * 4);
  float* bufA = (float*)alloc((size_t)NBV * DD * 4);   // tmp / H
  float* bufB = (float*)alloc((size_t)NBV * DD * 4);   // agg (f32)
  float* Qbuf = (float*)alloc((size_t)NLT * DD * 4);
  float* kkb  = (float*)alloc((size_t)NLT * DD * 4);
  float* vvb  = (float*)alloc((size_t)NLT * DD * 4);
  float* degA[4];
  degA[0] = (float*)alloc((size_t)NBV * 4);
  degA[1] = (float*)alloc((size_t)NB  * 4);
  degA[2] = (float*)alloc((size_t)NLT * 4);
  degA[3] = (float*)alloc((size_t)NB  * 4);
  float* gvec = (float*)alloc((size_t)NBV * 4);

  // bf16 shadows (sh_cur[t] aliased across layers: cur-shadow is dead once
  // the layer's relations finish, then reused for the new-shadow)
  unsigned short* sh_cur[4];
  for (int t = 0; t < 4; ++t) sh_cur[t] = (unsigned short*)alloc((size_t)Nn[t] * DD * 2);
  unsigned short* sh_agg  = (unsigned short*)alloc((size_t)NBV * DD * 2);
  unsigned short* proj_bf = (unsigned short*)alloc((size_t)NBV * DD * 2);
  unsigned short* sh_T    = (unsigned short*)alloc((size_t)NLT * DD * 2);
  unsigned short* sh_tt   = (unsigned short*)alloc((size_t)78 * 64 * 2);

  unsigned short* wsrc_p = (unsigned short*)alloc((size_t)8 * 16384 * 2);
  unsigned short* wdst_p = (unsigned short*)alloc((size_t)8 * 16384 * 2);
  unsigned short* wm_p   = (unsigned short*)alloc((size_t)8 * 16384 * 2);
  unsigned short* wtp_p  = (unsigned short*)alloc((size_t)8192  * 2);
  unsigned short* wq_p   = (unsigned short*)alloc((size_t)16384 * 2);
  unsigned short* wk_p   = (unsigned short*)alloc((size_t)16384 * 2);
  unsigned short* wv_p   = (unsigned short*)alloc((size_t)16384 * 2);
  unsigned short* wa_p   = (unsigned short*)alloc((size_t)16384 * 2);
  unsigned short* wb_p   = (unsigned short*)alloc((size_t)16384 * 2);

  auto zero = [&](float* p, long n) {
    fill_zero_kernel<<<(unsigned)((n + 255) / 256), 256, 0, stream>>>(p, n);
  };
  auto cvt = [&](const float* src, unsigned short* dst, long n) {
    long n4 = n / 4;
    cvt_bf16_kernel<<<(unsigned)((n4 + 255) / 256), 256, 0, stream>>>(src, dst, n4);
  };
  auto pack = [&](const float* W, const float* W2, unsigned short* o, int K) {
    pack_w_kernel<<<(K * 128 + 255) / 256, 256, 0, stream>>>(W, W2, o, K);
  };

  // ---- weight packing (bf16, WMMA B-fragment order) ----
  for (int wi = 0; wi < 8; ++wi) {
    pack(W_src + (long)wi * 16384, nullptr, wsrc_p + (long)wi * 16384, 128);
    pack(W_dst + (long)wi * 16384, nullptr, wdst_p + (long)wi * 16384, 128);
    pack(W_m   + (long)wi * 16384, nullptr, wm_p   + (long)wi * 16384, 128);
  }
  pack(W_tp, nullptr, wtp_p, 64);
  pack(W_q, nullptr, wq_p, 128);
  pack(W_k, nullptr, wk_p, 128);
  pack(W_v, nullptr, wv_p, 128);
  pack(gW1, gW1 + 16384, wa_p, 128);      // concat[bv,bv,agg]: blocks 0+1 sum
  pack(gW1 + 32768, nullptr, wb_p, 128);

  // ---- bf16 shadows of the input node features + topic table ----
  for (int t = 0; t < 4; ++t) cvt(in_h[t], sh_cur[t], Nn[t] * DD);
  cvt(topic_table, sh_tt, 78 * 64);

  // ---- degrees (layer-invariant) ----
  const int*  esrc[4] = { ev, ei, es, esp };
  const long  Ecnt[4] = { 1000000, 200000, 30000, 300000 };
  const int   relS[4] = { 2, 0, 3, 2 };   // lt->bv, bv->b, leg->lt, lt->b
  const int   relT[4] = { 0, 1, 2, 1 };
  zero(degA[0], NBV); zero(degA[1], NB); zero(degA[2], NLT); zero(degA[3], NB);
  for (int r = 0; r < 4; ++r)
    deg_kernel<<<(unsigned)((Ecnt[r] + 255) / 256), 256, 0, stream>>>(
        esrc[r] + Ecnt[r], (int)Ecnt[r], degA[r]);

  // ---- T (bf16-only) and Q (layer-invariant) ----
  launch_gemm(stream, sh_tt, 64, topic_ids, wtp_p, 64, b_tp,
              nullptr, nullptr, 0, 0, 0, nullptr, sh_T, NLT);
  launch_gemm(stream, sh_T, 128, nullptr, wq_p, 128, b_q,
              nullptr, nullptr, 0, 0, 0, Qbuf, nullptr, NLT);

  // ---- 2 layers ----
  for (int l = 0; l < 2; ++l) {
    float* nxt[4];
    if (l == 0) { nxt[0] = new0[0]; nxt[1] = new0[1]; nxt[2] = new0[2]; nxt[3] = new0[3]; }
    else        { nxt[0] = new1_bv; nxt[1] = out_b;   nxt[2] = out_lt;  nxt[3] = out_leg; }
    for (int t = 0; t < 4; ++t) zero(nxt[t], Nn[t] * DD);

    for (int r = 0; r < 4; ++r) {
      const long Ns = Nn[relS[r]], Nd = Nn[relT[r]], E = Ecnt[r];
      const int wi = l * 4 + r;
      // proj = xs @ W_src[l,r]  (bf16-only output)
      launch_gemm(stream, sh_cur[relS[r]], 128, nullptr, wsrc_p + (long)wi * 16384,
                  128, nullptr, nullptr, nullptr, 0, 0, 0, nullptr, proj_bf, (int)Ns);
      // agg = segment_sum(proj[src], dst)/deg + b_src
      zero(bufB, Nd * DD);
      scatter_kernel<<<(unsigned)((E * 32 + 255) / 256), 256, 0, stream>>>(
          proj_bf, esrc[r], esrc[r] + E, E, bufB);
      normalize_kernel<<<(unsigned)((Nd * 32 + 255) / 256), 256, 0, stream>>>(
          bufB, sh_agg, degA[r], b_src + (long)wi * DD, Nd);
      // tmp = agg @ W_m + b_m
      launch_gemm(stream, sh_agg, 128, nullptr, wm_p + (long)wi * 16384, 128,
                  b_m + (long)wi * DD, nullptr, nullptr, 0, 0, 0, bufA, nullptr, (int)Nd);
      // new[t] += sigmoid(rel_attn[r]) * relu(xd @ W_dst + b_dst + tmp)
      launch_gemm(stream, sh_cur[relT[r]], 128, nullptr, wdst_p + (long)wi * 16384,
                  128, b_dst + (long)wi * DD, bufA, rel_attn, r, 1, 1,
                  nxt[relT[r]], nullptr, (int)Nd);
    }
    // (outer relu is identity: every contribution is sigmoid(.)*relu(.) >= 0)

    // finalize shadows for next consumers
    cvt(nxt[0], sh_cur[0], Nn[0] * DD);              // bv (layer1 or gate)
    if (l == 0) cvt(nxt[1], sh_cur[1], Nn[1] * DD);  // b (only needed for layer1)

    // topic attention for lt and leg
    for (int t = 2; t <= 3; ++t) {
      const int M = (int)Nn[t];
      cvt(nxt[t], sh_cur[t], Nn[t] * DD);            // pre-attn shadow for K/V
      launch_gemm(stream, sh_cur[t], 128, nullptr, wk_p, 128, b_k,
                  nullptr, nullptr, 0, 0, 0, kkb, nullptr, M);
      launch_gemm(stream, sh_cur[t], 128, nullptr, wv_p, 128, b_v,
                  nullptr, nullptr, 0, 0, 0, vvb, nullptr, M);
      attn_kernel<<<(M + 7) / 8, 256, 0, stream>>>(Qbuf, kkb, vvb, nxt[t],
                                                   sh_cur[t], M);   // post-attn shadow
    }
  }

  // ---- final bv gate: e_isver[0]==arange -> segment mean is a row gather ----
  gather_rows_kernel<<<(unsigned)(((long)NBV * 32 + 255) / 256), 256, 0, stream>>>(
      out_b, ei + 200000, bufB, sh_agg, NBV);        // agg = b[e1], f32 + bf16
  // H = bv@(gW1_0+gW1_1) + gb1
  launch_gemm(stream, sh_cur[0], 128, nullptr, wa_p, 128, gb1,
              nullptr, nullptr, 0, 0, 0, bufA, nullptr, NBV);
  // H = relu(H + agg@gW1_2)
  launch_gemm(stream, sh_agg, 128, nullptr, wb_p, 128, nullptr,
              bufA, nullptr, 0, 1, 0, bufA, nullptr, NBV);
  rowdot_sig_kernel<<<(NBV + 7) / 8, 256, 0, stream>>>(bufA, gW2, gb2, gvec, NBV);
  gateout_kernel<<<(unsigned)(((long)NBV * 32 + 255) / 256), 256, 0, stream>>>(
      gvec, bufB, new1_bv, out_bv, NBV);
}
